// VQVAE_1494648619467
// MI455X (gfx1250) — matrix operand order repaired and checked by reading, not compiled, not run on hardware
//
#include <hip/hip_runtime.h>
#include <stdint.h>

// ---------------------------------------------------------------------------
// Types for CDNA5 WMMA (wave32).  v16bf = 8 VGPRs of bf16 A/B fragment,
// v8f = 8 VGPRs of f32 C/D fragment.
// ---------------------------------------------------------------------------
typedef __attribute__((ext_vector_type(16))) __bf16         v16bf;
typedef __attribute__((ext_vector_type(8)))  float          v8f;
typedef __attribute__((ext_vector_type(8)))  unsigned short v8u;

union Frag {
    v16bf bf;
    v8u   hv[2];   // two 16-byte chunks per lane
};

__device__ __forceinline__ unsigned short f2bf(float f) {
    unsigned int u = __float_as_uint(f);
    unsigned int r = u + 0x7FFFu + ((u >> 16) & 1u);   // round-to-nearest-even
    return (unsigned short)(r >> 16);
}

__device__ __forceinline__ v8f wmma_bf16(const Frag& a, const Frag& b, v8f c) {
    return __builtin_amdgcn_wmma_f32_16x16x32_bf16(
        /*neg_a=*/false, a.bf, /*neg_b=*/false, b.bf,
        /*c_mod=*/(short)0, c, /*reuse_a=*/false, /*reuse_b=*/false);
}

// ---------------------------------------------------------------------------
// CDNA5 async global->LDS copy (16B per lane, tracked by ASYNCcnt).
// LDS address operand is a byte offset within the wave's LDS allocation;
// generic pointers to __shared__ carry it in their low 32 bits.
// ---------------------------------------------------------------------------
__device__ __forceinline__ void async_ld16(const void* gsrc, void* lds_dst) {
    unsigned           d = (unsigned)(uintptr_t)lds_dst;
    unsigned long long a = (unsigned long long)(uintptr_t)gsrc;
    asm volatile("global_load_async_to_lds_b128 %0, %1, off"
                 :: "v"(d), "v"(a) : "memory");
}
__device__ __forceinline__ void wait_async0() {
    asm volatile("s_wait_asynccnt 0x0" ::: "memory");
}

// ---------------------------------------------------------------------------
// Problem dimensions (from setup_inputs)
// ---------------------------------------------------------------------------
#define NB   32
#define H0   256
#define H1   128
#define H2   64
#define KCB  512
#define M1   (NB*H1*H1)   // 524288
#define M2   (NB*H2*H2)   // 131072
#define MD2  (NB*H1*H1)   // 524288  (per-parity-class rows of deconv2)
#define RECON_ELEMS (NB*3*H0*H0)  // 6291456

// ---------------------------------------------------------------------------
// Generic implicit-GEMM WMMA kernel, double-buffered async pipeline.
//   C[M x N] (f32, consumed by epilogue functor) = A[M x K] * Bt[N x Kpad]^T
// Block = 256 threads = 8 wave32s.  Block tile: BM = WR*32, BN = WC*16*NSUB.
// LDS row stride of 40 halfs (80B) keeps every 16B fragment chunk aligned.
// Pipeline: issue async DMA for tile i+1, run WMMA on tile i, then
// s_wait_asynccnt 0 + one barrier per k-step (ping-pong LDS buffers).
// ---------------------------------------------------------------------------
template <typename GA, typename EPI, int WR, int WC, int NSUB>
__global__ __launch_bounds__(256) void gemm_wmma_kernel(
        GA ga, const unsigned short* __restrict__ Bt, EPI epi,
        int N, int Kpad)
{
    constexpr int BM  = WR * 32;
    constexpr int BN  = WC * 16 * NSUB;
    constexpr int LDT = 40;                       // halfs per LDS row

    __shared__ unsigned short lds_a[2][BM * LDT];
    __shared__ unsigned short lds_b[2][BN * LDT];

    const int tid  = threadIdx.x;
    const int nbn  = N / BN;
    const int bm   = (blockIdx.x / nbn) * BM;
    const int bn   = (blockIdx.x % nbn) * BN;

    const int wave = tid >> 5;
    const int lane = tid & 31;
    const int hfl  = lane >> 4;        // 0 = lanes 0-15, 1 = lanes 16-31
    const int l16  = lane & 15;
    const int wr   = wave % WR;
    const int wc   = wave / WR;

    // ---- tile staging (async DMA into LDS; zero-fill for padding rows) ----
    auto stageA = [&](unsigned short* la, int k0) {
        if constexpr (GA::kVector) {
#pragma unroll
            for (int e = tid; e < BM * 2; e += 256) {
                int r = e >> 1, seg = (e & 1) * 16;
                unsigned short* dst = &la[r * LDT + seg];
                const unsigned short* src = ga.row16(bm + r, k0 + seg);
                if (src) {
                    async_ld16(src,     dst);
                    async_ld16(src + 8, dst + 8);
                } else {
                    v8u z;
#pragma unroll
                    for (int j = 0; j < 8; ++j) z[j] = 0;
                    *(v8u*)dst = z;
                    *(v8u*)(dst + 8) = z;
                }
            }
        } else {
#pragma unroll
            for (int e = tid; e < BM * 32; e += 256) {
                int r = e >> 5, kk = e & 31;
                la[r * LDT + kk] = ga(bm + r, k0 + kk);
            }
        }
    };
    auto stageB = [&](unsigned short* lb, int k0) {
#pragma unroll
        for (int e = tid; e < BN * 2; e += 256) {
            int r = e >> 1, seg = (e & 1) * 16;
            const unsigned short* src = Bt + (size_t)(bn + r) * Kpad + k0 + seg;
            unsigned short* dst = &lb[r * LDT + seg];
            async_ld16(src,     dst);
            async_ld16(src + 8, dst + 8);
        }
    };

    v8f acc[2][NSUB];
#pragma unroll
    for (int s = 0; s < 2; ++s)
#pragma unroll
        for (int t = 0; t < NSUB; ++t)
#pragma unroll
            for (int j = 0; j < 8; ++j) acc[s][t][j] = 0.0f;

    // ---- prologue: stage first tile ----
    stageA(lds_a[0], 0);
    stageB(lds_b[0], 0);
    wait_async0();
    __syncthreads();

    int buf = 0;
    for (int k0 = 0; k0 < Kpad; k0 += 32) {
        const bool hasNext = (k0 + 32) < Kpad;
        if (hasNext) {                         // kick DMA for next tile
            stageA(lds_a[buf ^ 1], k0 + 32);
            stageB(lds_b[buf ^ 1], k0 + 32);
        }
        const unsigned short* la = lds_a[buf];
        const unsigned short* lb = lds_b[buf];

        // ---- build fragments (ISA 16-bit A / B VGPR layouts) ----
        Frag a[2], b[NSUB];
#pragma unroll
        for (int s = 0; s < 2; ++s) {
            int row = wr * 32 + s * 16 + l16;
            a[s].hv[0] = *(const v8u*)&la[row * LDT + hfl * 8];
            a[s].hv[1] = *(const v8u*)&la[row * LDT + hfl * 8 + 16];
        }
#pragma unroll
        for (int t = 0; t < NSUB; ++t) {
            int col = wc * 16 * NSUB + t * 16 + l16;
            b[t].hv[0] = *(const v8u*)&lb[col * LDT + hfl * 8];
            b[t].hv[1] = *(const v8u*)&lb[col * LDT + hfl * 8 + 16];
        }
#pragma unroll
        for (int s = 0; s < 2; ++s)
#pragma unroll
            for (int t = 0; t < NSUB; ++t)
                acc[s][t] = wmma_bf16(a[s], b[t], acc[s][t]);

        if (hasNext) {                         // retire next tile's DMA
            wait_async0();
            __syncthreads();
            buf ^= 1;
        }
    }

    // ---- epilogue: C-tile layout is (row = i + 8*hfl, col = l16) ----
#pragma unroll
    for (int s = 0; s < 2; ++s)
#pragma unroll
        for (int t = 0; t < NSUB; ++t) {
            int mt = bm + wr * 32 + s * 16 + hfl * 8;
            int nt = bn + wc * 16 * NSUB + t * 16 + l16;
#pragma unroll
            for (int i = 0; i < 8; ++i)
                epi(mt + i, nt, acc[s][t][i]);
        }
}

// ---------------------------------------------------------------------------
// Gather functors.  Vector gathers return a pointer to 16 contiguous bf16
// (channel-fastest im2col) or nullptr for padding rows.
// ---------------------------------------------------------------------------
struct GConv1 {                 // x NHWC bf16 [32][256][256][3], k=(ky*4+kx)*3+c
    static constexpr bool kVector = false;
    const unsigned short* xb;
    __device__ unsigned short operator()(int m, int k) const {
        if (k >= 48) return 0;
        int n = m >> 14, rem = m & 16383, u = rem >> 7, v = rem & 127;
        int t = k / 3, c = k - 3 * t;
        int ky = t >> 2, kx = t & 3;
        int iy = 2 * u - 1 + ky, ix = 2 * v - 1 + kx;
        if ((unsigned)iy >= 256u || (unsigned)ix >= 256u) return 0;
        return xb[(((size_t)n * 256 + iy) * 256 + ix) * 3 + c];
    }
};
struct GConv2 {                 // z1 NHWC bf16 [32][128][128][128]
    static constexpr bool kVector = true;
    const unsigned short* z1;
    __device__ const unsigned short* row16(int m, int k) const {
        int n = m >> 12, rem = m & 4095, u = rem >> 6, v = rem & 63;
        int t = k >> 7, ci = k & 127;
        int ky = t >> 2, kx = t & 3;
        int iy = 2 * u - 1 + ky, ix = 2 * v - 1 + kx;
        if ((unsigned)iy >= 128u || (unsigned)ix >= 128u) return nullptr;
        return z1 + (((size_t)n * 128 + iy) * 128 + ix) * 128 + ci;
    }
};
struct GDec1 {                  // q NHWC bf16 [32][64][64][64], k=(ty*2+tx)*64+ci
    static constexpr bool kVector = true;
    const unsigned short* q;
    int dy0, dy1, dx0, dx1;
    __device__ const unsigned short* row16(int m, int k) const {
        int n = m >> 12, rem = m & 4095, u = rem >> 6, v = rem & 63;
        int t = k >> 6, ci = k & 63;
        int ty = (t >> 1) & 1, tx = t & 1;
        int iy = u + (ty ? dy1 : dy0);
        int ix = v + (tx ? dx1 : dx0);
        if ((unsigned)iy >= 64u || (unsigned)ix >= 64u) return nullptr;
        return q + (((size_t)n * 64 + iy) * 64 + ix) * 64 + ci;
    }
};
struct GDec2 {                  // h NHWC bf16 [32][128][128][128], k=(ty*2+tx)*128+ci
    static constexpr bool kVector = true;
    const unsigned short* h;
    int dy0, dy1, dx0, dx1;
    __device__ const unsigned short* row16(int m, int k) const {
        int n = m >> 14, rem = m & 16383, u = rem >> 7, v = rem & 127;
        int t = k >> 7, ci = k & 127;
        int ty = (t >> 1) & 1, tx = t & 1;
        int iy = u + (ty ? dy1 : dy0);
        int ix = v + (tx ? dx1 : dx0);
        if ((unsigned)iy >= 128u || (unsigned)ix >= 128u) return nullptr;
        return h + (((size_t)n * 128 + iy) * 128 + ix) * 128 + ci;
    }
};

// ---------------------------------------------------------------------------
// Epilogue functors
// ---------------------------------------------------------------------------
struct EpiConv1 {               // relu(+bias) -> z1 NHWC bf16 (m is flat NHW)
    unsigned short* z1; const float* bias;
    __device__ void operator()(int m, int n, float v) const {
        z1[(size_t)m * 128 + n] = f2bf(fmaxf(v + bias[n], 0.0f));
    }
};
struct EpiConv2 {               // +bias -> z f32 and bf16 copies (flat [M2][64])
    float* zf; unsigned short* fb; const float* bias;
    __device__ void operator()(int m, int n, float v) const {
        float r = v + bias[n];
        zf[(size_t)m * 64 + n] = r;
        fb[(size_t)m * 64 + n] = f2bf(r);
    }
};
struct EpiDec1 {                // relu(+bias) -> h NHWC bf16 at parity class site
    unsigned short* h; const float* bias; int py, px;
    __device__ void operator()(int m, int n, float v) const {
        int nn = m >> 12, rem = m & 4095, u = rem >> 6, vv = rem & 63;
        int oy = 2 * u + py, ox = 2 * vv + px;
        h[(((size_t)nn * 128 + oy) * 128 + ox) * 128 + n] =
            f2bf(fmaxf(v + bias[n], 0.0f));
    }
};
struct EpiDec2 {                // tanh(+bias) -> recon NCHW f32 (cols >= 3 padded)
    float* out; const float* bias; int py, px;
    __device__ void operator()(int m, int n, float v) const {
        if (n >= 3) return;
        int nn = m >> 14, rem = m & 16383, u = rem >> 7, vv = rem & 127;
        int oy = 2 * u + py, ox = 2 * vv + px;
        out[(((size_t)nn * 3 + n) * 256 + oy) * 256 + ox] = tanhf(v + bias[n]);
    }
};

// ---------------------------------------------------------------------------
// VQ nearest-neighbor: scores = ||c||^2 - 2 z.c^T via WMMA, per-row argmin.
// ---------------------------------------------------------------------------
__global__ __launch_bounds__(256) void vq_argmin_kernel(
        const unsigned short* __restrict__ flat,   // [M2][64] bf16
        const unsigned short* __restrict__ cbb,    // [512][64] bf16
        const float* __restrict__ cbn,             // [512] ||c||^2
        int* __restrict__ idx)                     // [M2]
{
    const int tid  = threadIdx.x;
    const int wave = tid >> 5, lane = tid & 31;
    const int hfl  = lane >> 4, l16 = lane & 15;
    const int mbase = (blockIdx.x * 8 + wave) * 16;

    Frag a0, a1;
    const unsigned short* ar = flat + (size_t)(mbase + l16) * 64;
    a0.hv[0] = *(const v8u*)(ar + hfl * 8);
    a0.hv[1] = *(const v8u*)(ar + hfl * 8 + 16);
    a1.hv[0] = *(const v8u*)(ar + 32 + hfl * 8);
    a1.hv[1] = *(const v8u*)(ar + 32 + hfl * 8 + 16);

    float mval[8]; int midx[8];
#pragma unroll
    for (int i = 0; i < 8; ++i) { mval[i] = 3.4e38f; midx[i] = 0; }

    for (int nt = 0; nt < 32; ++nt) {
        Frag b0, b1;
        const unsigned short* br = cbb + (size_t)(nt * 16 + l16) * 64;
        b0.hv[0] = *(const v8u*)(br + hfl * 8);
        b0.hv[1] = *(const v8u*)(br + hfl * 8 + 16);
        b1.hv[0] = *(const v8u*)(br + 32 + hfl * 8);
        b1.hv[1] = *(const v8u*)(br + 32 + hfl * 8 + 16);

        v8f acc;
#pragma unroll
        for (int j = 0; j < 8; ++j) acc[j] = 0.0f;
        acc = wmma_bf16(a0, b0, acc);
        acc = wmma_bf16(a1, b1, acc);

        int   col = nt * 16 + l16;
        float cn  = cbn[col];
#pragma unroll
        for (int i = 0; i < 8; ++i) {
            float val = cn - 2.0f * acc[i];
            if (val < mval[i]) { mval[i] = val; midx[i] = col; }
        }
    }
#pragma unroll
    for (int i = 0; i < 8; ++i) {
#pragma unroll
        for (int d = 1; d < 16; d <<= 1) {
            float ov = __shfl_xor(mval[i], d, 32);
            int   oi = __shfl_xor(midx[i], d, 32);
            if (ov < mval[i] || (ov == mval[i] && oi < midx[i])) {
                mval[i] = ov; midx[i] = oi;
            }
        }
        if (l16 == 0) idx[mbase + hfl * 8 + i] = midx[i];
    }
}

// gather q = codebook[idx], accumulate (q-z)^2, emit q as bf16 for the decoder
__global__ __launch_bounds__(256) void vq_finalize_kernel(
        const int* __restrict__ idx, const float* __restrict__ cb,
        const float* __restrict__ zf, unsigned short* __restrict__ qb,
        float* __restrict__ lossAcc)
{
    __shared__ float red[256];
    int m = blockIdx.x * 256 + threadIdx.x;
    float s = 0.0f;
    if (m < M2) {
        int e = idx[m];
        const float* cr = cb + (size_t)e * 64;
        const float* zr = zf + (size_t)m * 64;
        unsigned short* qr = qb + (size_t)m * 64;
#pragma unroll 4
        for (int d = 0; d < 64; ++d) {
            float q = cr[d], df = q - zr[d];
            s += df * df;
            qr[d] = f2bf(q);
        }
    }
    red[threadIdx.x] = s;
    __syncthreads();
    for (int st = 128; st > 0; st >>= 1) {
        if (threadIdx.x < st) red[threadIdx.x] += red[threadIdx.x + st];
        __syncthreads();
    }
    if (threadIdx.x == 0) atomicAdd(lossAcc, red[0]);
}

__global__ void zero_loss_kernel(float* loss) { *loss = 0.0f; }
__global__ void write_loss_kernel(const float* loss, float* out) {
    // e_latent == q_latent in forward; vq_loss = (1 + 0.25) * mean((q-z)^2)
    out[RECON_ELEMS] = 1.25f * (*loss) / (float)(M2 * 64);
}

// ---------------------------------------------------------------------------
// Packing kernels (f32 -> bf16 layout transforms; tiny vs. the GEMMs)
// ---------------------------------------------------------------------------
__global__ void pack_x_kernel(const float* __restrict__ x,
                              unsigned short* __restrict__ xb) {
    int e = blockIdx.x * 256 + threadIdx.x;
    if (e >= NB * 256 * 256 * 3) return;
    int c = e % 3, t = e / 3;
    int w = t % 256; t /= 256;
    int h = t % 256; int n = t / 256;
    xb[e] = f2bf(x[(((size_t)n * 3 + c) * 256 + h) * 256 + w]);
}
__global__ void pack_w1_kernel(const float* __restrict__ w,
                               unsigned short* __restrict__ Bt) {
    int e = blockIdx.x * 256 + threadIdx.x;
    if (e >= 128 * 64) return;
    int k = e & 63, n = e >> 6;
    unsigned short v = 0;
    if (k < 48) {
        int t = k / 3, c = k - 3 * t, ky = t >> 2, kx = t & 3;
        v = f2bf(w[(((size_t)n * 3 + c) * 4 + ky) * 4 + kx]);
    }
    Bt[e] = v;
}
__global__ void pack_w2_kernel(const float* __restrict__ w,
                               unsigned short* __restrict__ Bt) {
    int e = blockIdx.x * 256 + threadIdx.x;
    if (e >= 64 * 2048) return;
    int k = e & 2047, n = e >> 11;
    int t = k >> 7, ci = k & 127, ky = t >> 2, kx = t & 3;
    Bt[e] = f2bf(w[(((size_t)n * 128 + ci) * 4 + ky) * 4 + kx]);
}
__global__ void pack_cb_kernel(const float* __restrict__ cb,
                               unsigned short* __restrict__ cbb) {
    int e = blockIdx.x * 256 + threadIdx.x;
    if (e < KCB * 64) cbb[e] = f2bf(cb[e]);
}
__global__ void pack_cbnorm_kernel(const float* __restrict__ cb,
                                   float* __restrict__ cbn) {
    int e = blockIdx.x * 256 + threadIdx.x;
    if (e >= KCB) return;
    float s = 0.0f;
    for (int d = 0; d < 64; ++d) { float v = cb[(size_t)e * 64 + d]; s += v * v; }
    cbn[e] = s;
}
__device__ __forceinline__ int tapk(int p, int t) {   // parity/tap -> kernel idx
    return p ? (t ? 2 : 0) : (t ? 3 : 1);
}
__global__ void pack_d1_kernel(const float* __restrict__ w,  // (64,128,4,4)
                               unsigned short* __restrict__ Bt) {
    int e = blockIdx.x * 256 + threadIdx.x;
    if (e >= 4 * 128 * 256) return;
    int cls = e >> 15, rem = e & 32767, co = rem >> 8, k = rem & 255;
    int py = cls >> 1, px = cls & 1;
    int t = k >> 6, ci = k & 63, ty = (t >> 1) & 1, tx = t & 1;
    int ky = tapk(py, ty), kx = tapk(px, tx);
    Bt[e] = f2bf(w[(((size_t)ci * 128 + co) * 4 + ky) * 4 + kx]);
}
__global__ void pack_d2_kernel(const float* __restrict__ w,  // (128,3,4,4)
                               unsigned short* __restrict__ Bt) {
    int e = blockIdx.x * 256 + threadIdx.x;
    if (e >= 4 * 16 * 512) return;
    int cls = e >> 13, rem = e & 8191, co = rem >> 9, k = rem & 511;
    int py = cls >> 1, px = cls & 1;
    int t = k >> 7, ci = k & 127, ty = (t >> 1) & 1, tx = t & 1;
    int ky = tapk(py, ty), kx = tapk(px, tx);
    Bt[e] = (co < 3) ? f2bf(w[(((size_t)ci * 3 + co) * 4 + ky) * 4 + kx])
                     : (unsigned short)0;
}

// ---------------------------------------------------------------------------
// Launch
// ---------------------------------------------------------------------------
extern "C" void kernel_launch(void* const* d_in, const int* in_sizes, int n_in,
                              void* d_out, int out_size, void* d_ws, size_t ws_size,
                              hipStream_t stream) {
    (void)in_sizes; (void)n_in; (void)out_size; (void)ws_size;
    const float* x      = (const float*)d_in[0];
    const float* enc_w1 = (const float*)d_in[1];
    const float* enc_b1 = (const float*)d_in[2];
    const float* enc_w2 = (const float*)d_in[3];
    const float* enc_b2 = (const float*)d_in[4];
    const float* cbk    = (const float*)d_in[5];
    const float* dec_w1 = (const float*)d_in[6];
    const float* dec_b1 = (const float*)d_in[7];
    const float* dec_w2 = (const float*)d_in[8];
    const float* dec_b2 = (const float*)d_in[9];
    float* out = (float*)d_out;

    // -------- workspace carve (256B aligned), total ~336 MB --------
    uintptr_t p = (uintptr_t)d_ws;
    auto take = [&](size_t bytes) -> void* {
        void* r = (void*)p;
        p += (bytes + 255) & ~(size_t)255;
        return r;
    };
    unsigned short* xb   = (unsigned short*)take((size_t)NB*256*256*3*2);
    unsigned short* Bt1  = (unsigned short*)take((size_t)128*64*2);
    unsigned short* Bt2  = (unsigned short*)take((size_t)64*2048*2);
    unsigned short* cbb  = (unsigned short*)take((size_t)KCB*64*2);
    float*          cbn  = (float*)take((size_t)KCB*4);
    unsigned short* Btd1 = (unsigned short*)take((size_t)4*128*256*2);
    unsigned short* Btd2 = (unsigned short*)take((size_t)4*16*512*2);
    unsigned short* z1b  = (unsigned short*)take((size_t)M1*128*2);   // 128 MB
    float*          zf   = (float*)take((size_t)M2*64*4);             // 33.5 MB
    unsigned short* fb   = (unsigned short*)take((size_t)M2*64*2);
    int*            idx  = (int*)take((size_t)M2*4);
    unsigned short* qb   = (unsigned short*)take((size_t)M2*64*2);
    unsigned short* hb   = (unsigned short*)take((size_t)M1*128*2);   // 128 MB
    float*          loss = (float*)take(4);

    // -------- packing --------
    pack_x_kernel <<<(NB*256*256*3 + 255)/256, 256, 0, stream>>>(x, xb);
    pack_w1_kernel<<<(128*64 + 255)/256,       256, 0, stream>>>(enc_w1, Bt1);
    pack_w2_kernel<<<(64*2048 + 255)/256,      256, 0, stream>>>(enc_w2, Bt2);
    pack_cb_kernel<<<(KCB*64 + 255)/256,       256, 0, stream>>>(cbk, cbb);
    pack_cbnorm_kernel<<<(KCB + 255)/256,      256, 0, stream>>>(cbk, cbn);
    pack_d1_kernel<<<(4*128*256 + 255)/256,    256, 0, stream>>>(dec_w1, Btd1);
    pack_d2_kernel<<<(4*16*512 + 255)/256,     256, 0, stream>>>(dec_w2, Btd2);

    // -------- encoder conv1: M=524288, N=128, K=48 (pad 64), tile 128x128 ----
    {
        GConv1 g{xb};
        EpiConv1 e{z1b, enc_b1};
        gemm_wmma_kernel<GConv1, EpiConv1, 4, 2, 4>
            <<<(M1/128)*(128/128), 256, 0, stream>>>(g, Bt1, e, 128, 64);
    }
    // -------- encoder conv2: M=131072, N=64, K=2048, tile 128x64 --------
    {
        GConv2 g{z1b};
        EpiConv2 e{zf, fb, enc_b2};
        gemm_wmma_kernel<GConv2, EpiConv2, 4, 2, 2>
            <<<(M2/128)*(64/64), 256, 0, stream>>>(g, Bt2, e, 64, 2048);
    }
    // -------- VQ: distances + argmin + loss + quantized gather --------
    vq_argmin_kernel<<<M2/128, 256, 0, stream>>>(fb, cbb, cbn, idx);
    zero_loss_kernel<<<1, 1, 0, stream>>>(loss);
    vq_finalize_kernel<<<(M2 + 255)/256, 256, 0, stream>>>(idx, cbk, zf, qb, loss);

    // -------- decoder deconv1 (64->128), 4 parity classes, tile 128x128 -----
    for (int cls = 0; cls < 4; ++cls) {
        int py = cls >> 1, px = cls & 1;
        GDec1 g{qb, py ? 1 : 0, py ? 0 : -1, px ? 1 : 0, px ? 0 : -1};
        EpiDec1 e{hb, dec_b1, py, px};
        gemm_wmma_kernel<GDec1, EpiDec1, 4, 2, 4>
            <<<(M2/128)*(128/128), 256, 0, stream>>>(
                g, Btd1 + (size_t)cls*128*256, e, 128, 256);
    }
    // -------- decoder deconv2 (128->3), 4 parity classes, tile 256x16 -------
    for (int cls = 0; cls < 4; ++cls) {
        int py = cls >> 1, px = cls & 1;
        GDec2 g{hb, py ? 1 : 0, py ? 0 : -1, px ? 1 : 0, px ? 0 : -1};
        EpiDec2 e{out, dec_b2, py, px};
        gemm_wmma_kernel<GDec2, EpiDec2, 8, 1, 1>
            <<<(MD2/256)*(16/16), 256, 0, stream>>>(
                g, Btd2 + (size_t)cls*16*512, e, 16, 512);
    }

    write_loss_kernel<<<1, 1, 0, stream>>>(loss, out);
}
